// ProtFill_2353642078945
// MI455X (gfx1250) — matrix operand order repaired and checked by reading, not compiled, not run on hardware
//
#include <hip/hip_runtime.h>
#include <hip/hip_bf16.h>
#include <hip/hip_fp16.h>

// ---------------------------------------------------------------------------
// ProteinMPNN edge featurization for MI455X (gfx1250, wave32, WMMA).
//   K1 atoms_kernel : build [N, Ca, Cb, C, O] per residue
//   K2 wprep_kernel : permute w_edge (416x128 f32) -> f16 B-fragment image
//   K3 topk_kernel  : masked pairwise dist + top-30 (shfl-based argmin)
//   K4 edge_kernel  : async-DMA weights -> LDS (overlapped with RBF/pos
//                     feature build) -> f16 WMMA GEMM -> LayerNorm -> store
// ---------------------------------------------------------------------------

#define NRES  2048
#define BATCH 2
#define TOPK  30
#define KDIM  416      // 16 pos + 25*16 RBF
#define NDIM  128
#define NEDGE (BATCH * NRES * TOPK)          // 122880
#define NMTILE (NEDGE / 16)                  // 7680 (exact)

#ifndef USE_ASYNC_STAGING
#define USE_ASYNC_STAGING 1
#endif

typedef _Float16 v16h __attribute__((ext_vector_type(16)));
typedef float    v8f  __attribute__((ext_vector_type(8)));

__device__ __constant__ int PA[24] = {0,1,2,3,3,3,0,0,2,0,1,2,1,2,1,4,3,4,0,4,2,4,4,1};
__device__ __constant__ int PB[24] = {0,1,2,0,1,2,1,2,1,3,3,3,0,0,2,4,4,3,4,0,4,1,2,4};

// ---------------------------------------------------------------- K1: atoms
__global__ __launch_bounds__(256) void atoms_kernel(const float* __restrict__ X,
                                                    float* __restrict__ atoms) {
  int idx = blockIdx.x * blockDim.x + threadIdx.x;   // b*NRES + i
  if (idx >= BATCH * NRES) return;
  const float* x = X + (size_t)idx * 12;             // [4 atoms][3]
  float N0=x[0],N1=x[1],N2=x[2];                     // X[...,0,:]
  float C0=x[3],C1=x[4],C2=x[5];                     // X[...,1,:]
  float A0=x[6],A1=x[7],A2=x[8];                     // Ca = X[...,2,:]
  float O0=x[9],O1=x[10],O2=x[11];                   // X[...,3,:]
  float b0=A0-N0, b1=A1-N1, b2=A2-N2;                // b = Ca - N
  float c0=C0-A0, c1=C1-A1, c2=C2-A2;                // c = C - Ca
  float a0=b1*c2-b2*c1, a1=b2*c0-b0*c2, a2=b0*c1-b1*c0;   // a = b x c
  float Cb0 = -0.58273431f*a0 + 0.56802827f*b0 - 0.54067466f*c0 + A0;
  float Cb1 = -0.58273431f*a1 + 0.56802827f*b1 - 0.54067466f*c1 + A1;
  float Cb2 = -0.58273431f*a2 + 0.56802827f*b2 - 0.54067466f*c2 + A2;
  float* o = atoms + (size_t)idx * 15;               // [N,Ca,Cb,C,O][3]
  o[0]=N0;  o[1]=N1;  o[2]=N2;
  o[3]=A0;  o[4]=A1;  o[5]=A2;
  o[6]=Cb0; o[7]=Cb1; o[8]=Cb2;
  o[9]=C0;  o[10]=C1; o[11]=C2;
  o[12]=O0; o[13]=O1; o[14]=O2;
}

// --------------------------------------------------- K2: weight f16 permute
// wperm[h] with h = (((t*8+nt)*32+lane)*16+e) holds
//   W[32t + (e<8 ? (lane>>4)*8 + e : 16 + (lane>>4)*8 + (e-8))][nt*16 + (lane&15)]  (B-fragment order)
#define LDSW_HALF (13*8*32*16)         // 53248 halves
__global__ __launch_bounds__(256) void wprep_kernel(const float* __restrict__ w_edge,
                                                    _Float16* __restrict__ wperm) {
  int h = blockIdx.x * 256 + threadIdx.x;
  if (h >= LDSW_HALF) return;
  int e = h & 15, ln = (h >> 4) & 31, nt = (h >> 9) & 7, t = h >> 12;
  int k = 32*t + ((e >> 3) << 4) + ((ln >> 4) << 3) + (e & 7);
  int n = nt*16 + (ln & 15);
  wperm[h] = (_Float16)w_edge[k * NDIM + n];
}

// ---------------------------------------------------------------- K3: top-k
// One 256-thread block per (b,i) row. Distance row in LDS; 30 argmin rounds
// on packed (dist_bits<<32 | j) keys (matches lax.top_k tie-break). Wave32
// shfl_xor reduction: 2 barriers per round instead of a full LDS tree.
__global__ __launch_bounds__(256) void topk_kernel(const float* __restrict__ X,
                                                   const float* __restrict__ mask,
                                                   int*  __restrict__ Eidx_out,
                                                   float* __restrict__ Dn_out) {
  __shared__ float dist[NRES];
  __shared__ unsigned long long wred[8];
  __shared__ float fred[8];
  int row  = blockIdx.x;                 // b*NRES + i
  int b    = row / NRES;
  int tid  = threadIdx.x;
  int lane = tid & 31, wav = tid >> 5;
  const float* Ci = X + ((size_t)row * 4 + 1) * 3;   // C atom (X[...,1,:])
  float cx = Ci[0], cy = Ci[1], cz = Ci[2];
  float mi = mask[row];
  float lmax = 0.0f;
  for (int j = tid; j < NRES; j += 256) {
    const float* Cj = X + (((size_t)b * NRES + j) * 4 + 1) * 3;
    float dx = Cj[0]-cx, dy = Cj[1]-cy, dz = Cj[2]-cz;
    float d  = sqrtf(dx*dx + dy*dy + dz*dz + 1e-6f);
    float m2 = mi * mask[b*NRES + j];
    float D  = m2 * d;
    dist[j]  = D;
    lmax = fmaxf(lmax, D);
  }
  for (int msk = 1; msk < 32; msk <<= 1) lmax = fmaxf(lmax, __shfl_xor(lmax, msk, 32));
  if (lane == 0) fred[wav] = lmax;
  __syncthreads();
  float Dmax = fred[0];
  for (int w = 1; w < 8; ++w) Dmax = fmaxf(Dmax, fred[w]);
  for (int j = tid; j < NRES; j += 256) {           // D_adj = D + (1-m2)*Dmax
    float m2 = mi * mask[b*NRES + j];
    dist[j] = dist[j] + (1.0f - m2) * Dmax;
  }
  __syncthreads();
  for (int it = 0; it < TOPK; ++it) {
    unsigned long long key = ~0ull;
    for (int j = tid; j < NRES; j += 256) {
      unsigned long long k =
          (((unsigned long long)__float_as_uint(dist[j])) << 32) | (unsigned)j;
      key = k < key ? k : key;
    }
    for (int msk = 1; msk < 32; msk <<= 1) {
      unsigned long long o = __shfl_xor(key, msk, 32);
      key = o < key ? o : key;
    }
    if (lane == 0) wred[wav] = key;
    __syncthreads();
    if (tid == 0) {
      unsigned long long k = wred[0];
      for (int w = 1; w < 8; ++w) k = wred[w] < k ? wred[w] : k;
      int   jb = (int)(k & 0xffffffffu);
      float dv = __uint_as_float((unsigned)(k >> 32));
      Eidx_out[row*TOPK + it] = jb;
      Dn_out  [row*TOPK + it] = dv;          // D_adj value, as in reference
      dist[jb] = __builtin_inff();
    }
    __syncthreads();
  }
}

// ------------------------------------------------------- K4: fused edge GEMM
// 256 threads = 8 waves. Each wave: one M-tile of 16 edges x full N=128.
// LDS: f16 weights (B-fragment image, 104KB, async-DMA'd) + per-wave
// A-fragment staging (13KB) + per-wave distance scratch (2KB) = 229376 B.
#define LDSA_OFF  (LDSW_HALF*2)        // 106496 B
#define LDSA_WAVE (13*32*16)           // 6656 halves / wave
#define EDATA_OFF (LDSA_OFF + 8*LDSA_WAVE*2)
#define SMEM_BYTES (EDATA_OFF + 8*2048)

__global__ __launch_bounds__(256) void edge_kernel(
    const float* __restrict__ atoms, const float* __restrict__ Dn,
    const int* __restrict__ Eidx, const int* __restrict__ resi,
    const int* __restrict__ chain, const float* __restrict__ w_pos,
    const float* __restrict__ b_pos, const _Float16* __restrict__ wperm,
    const float* __restrict__ gamma, const float* __restrict__ beta,
    float* __restrict__ Eout) {
  extern __shared__ char smem[];
  _Float16* lds_w = (_Float16*)smem;
  int tid  = threadIdx.x;
  int lane = tid & 31;
  int wav  = tid >> 5;

#if USE_ASYNC_STAGING
  // Kick off async DMA of the pre-permuted f16 weight image into LDS.
  // 53248 halves = 6656 x 16B chunks; 26 chunks per thread on ASYNCcnt.
  {
    unsigned ldsbase = (unsigned)(uintptr_t)smem;
    for (int c = tid; c < LDSW_HALF / 8; c += 256) {
      unsigned ldsa = ldsbase + (unsigned)c * 16u;
      const _Float16* gsrc = wperm + (size_t)c * 8;
      asm volatile("global_load_async_to_lds_b128 %0, %1, off"
                   :: "v"(ldsa), "v"(gsrc) : "memory");
    }
  }
#else
  for (int h = tid; h < LDSW_HALF; h += 256) lds_w[h] = wperm[h];
#endif

  int mt    = blockIdx.x * 8 + wav;          // M-tile (7680 total, exact)
  int edge0 = mt * 16;
  int b     = edge0 / (NRES * TOPK);         // uniform per tile (61440 % 16 == 0)
  float* myd = (float*)(smem + EDATA_OFF) + wav * 512;   // [16*25] dists
  int*   myj = (int*)(myd + 400);                        // [16] neighbor idx

  if (lane < 16) myj[lane] = Eidx[edge0 + lane];
  __syncthreads();

  // 25 distance channels per edge (overlaps with the weight DMA)
  for (int s = lane; s < 16*25; s += 32) {
    int m = s / 25, c = s % 25;
    int ed = edge0 + m;
    float d;
    if (c == 0) d = Dn[ed];
    else {
      int p = c - 1;
      int i = (ed / TOPK) % NRES;
      int j = myj[m];
      const float* Ai = atoms + (((size_t)b*NRES + i)*5 + PA[p]) * 3;
      const float* Bj = atoms + (((size_t)b*NRES + j)*5 + PB[p]) * 3;
      float dx = Ai[0]-Bj[0], dy = Ai[1]-Bj[1], dz = Ai[2]-Bj[2];
      d = sqrtf(dx*dx + dy*dy + dz*dz + 1e-6f);
    }
    myd[m*25 + c] = d;
  }
  __syncthreads();

  // Features -> LDS in A-fragment order.
  // f<16: positional embedding; else channel c=(f-16)>>4, rbf r=(f-16)&15.
  // A layout (16-bit 16x32): lane = g*16+m; e<8 -> k=g*8+e ; e>=8 -> k=16+g*8+(e-8)
  _Float16* mya = lds_w + LDSW_HALF + wav * LDSA_WAVE;
  for (int s = lane; s < 16*KDIM; s += 32) {
    int m = s / KDIM, f = s % KDIM;
    int ed = edge0 + m;
    float val;
    if (f < 16) {
      int i = (ed / TOPK) % NRES;
      int j = myj[m];
      int off = resi[b*NRES + i] - resi[b*NRES + j] + 32;
      off = off < 0 ? 0 : (off > 64 ? 64 : off);
      int dd = (chain[b*NRES + i] == chain[b*NRES + j]) ? off : 65;
      val = w_pos[dd*16 + f] + b_pos[f];
    } else {
      int c = (f - 16) >> 4, r = (f - 16) & 15;
      float mu = 2.0f + (20.0f / 15.0f) * (float)r;   // linspace(2,22,16)
      float u  = (myd[m*25 + c] - mu) * 0.8f;         // sigma = 1.25
      val = __expf(-u * u);
    }
    int t = f >> 5, kl = f & 31, g, e;
    if (kl < 16) { g = kl >> 3; e = kl & 7; }
    else         { int t2 = kl - 16; g = t2 >> 3; e = 8 + (t2 & 7); }
    mya[(t*32 + g*16 + m)*16 + e] = (_Float16)val;
  }

#if USE_ASYNC_STAGING
  asm volatile("s_wait_asynccnt 0x0" ::: "memory");   // weight DMA complete
#endif
  __syncthreads();

  // GEMM: 13 k-steps x 8 n-tiles of v_wmma_f32_16x16x32_f16, f32 accumulate.
  // All 8 B fragments of a k-step are loaded before the WMMA chain so the
  // scheduler can hoist ds_loads over the matrix ops.
  v8f acc[8] = {};
  for (int t = 0; t < 13; ++t) {
    v16h a = *(const v16h*)(mya + (t*32 + lane)*16);
    v16h bm[8];
#pragma unroll
    for (int nt = 0; nt < 8; ++nt)
      bm[nt] = *(const v16h*)(lds_w + ((t*8 + nt)*32 + lane)*16);
#pragma unroll
    for (int nt = 0; nt < 8; ++nt)
      acc[nt] = __builtin_amdgcn_wmma_f32_16x16x32_f16(
          false, a, false, bm[nt], (short)0, acc[nt], false, false);
  }

  // LayerNorm in registers. C/D layout: VGPR r -> row (lane<16 ? r : r+8),
  // n = nt*16 + (lane&15). Rows live within 16-lane halves -> xor-shuffles.
  float psum[8], psq[8];
  for (int r = 0; r < 8; ++r) {
    float s0 = 0.f, q0 = 0.f;
    for (int nt = 0; nt < 8; ++nt) { float v = acc[nt][r]; s0 += v; q0 += v*v; }
    psum[r] = s0; psq[r] = q0;
  }
  for (int msk = 1; msk < 16; msk <<= 1)
    for (int r = 0; r < 8; ++r) {
      psum[r] += __shfl_xor(psum[r], msk, 32);
      psq[r]  += __shfl_xor(psq[r],  msk, 32);
    }
  int nlo = lane & 15;
  float gam[8], bet[8];
  for (int nt = 0; nt < 8; ++nt) { gam[nt] = gamma[nt*16 + nlo]; bet[nt] = beta[nt*16 + nlo]; }
  for (int r = 0; r < 8; ++r) {
    float mean = psum[r] * (1.0f / 128.0f);
    float var  = psq[r] * (1.0f / 128.0f) - mean * mean;
    float rstd = rsqrtf(var + 1e-5f);
    int m  = (lane < 16) ? r : r + 8;
    float* orow = Eout + (size_t)(edge0 + m) * NDIM + nlo;
    for (int nt = 0; nt < 8; ++nt)
      orow[nt*16] = (acc[nt][r] - mean) * rstd * gam[nt] + bet[nt];
  }
}

// ---------------------------------------------------------------------------
extern "C" void kernel_launch(void* const* d_in, const int* in_sizes, int n_in,
                              void* d_out, int out_size, void* d_ws, size_t ws_size,
                              hipStream_t stream) {
  const float* X      = (const float*)d_in[0];
  const float* mask   = (const float*)d_in[1];
  const int*   resi   = (const int*)d_in[2];
  const int*   chain  = (const int*)d_in[3];
  const float* w_pos  = (const float*)d_in[4];
  const float* b_pos  = (const float*)d_in[5];
  const float* w_edge = (const float*)d_in[6];
  const float* gamma  = (const float*)d_in[7];
  const float* beta   = (const float*)d_in[8];

  float* Eout = (float*)d_out;                              // (B,N,30,128) f32
  int*   Eidx = (int*)d_out + (size_t)BATCH*NRES*TOPK*NDIM; // int32 region after E

  float*    atoms = (float*)d_ws;                               // 61440 f32
  float*    Dn    = atoms + (size_t)BATCH*NRES*15;              // 122880 f32
  _Float16* wperm = (_Float16*)((char*)d_ws + 737280);          // 53248 f16 (16B aligned)

  atoms_kernel<<<(BATCH*NRES + 255)/256, 256, 0, stream>>>(X, atoms);
  wprep_kernel<<<(LDSW_HALF + 255)/256, 256, 0, stream>>>(w_edge, wperm);
  topk_kernel <<<BATCH*NRES, 256, 0, stream>>>(X, mask, Eidx, Dn);
  edge_kernel <<<NMTILE/8, 256, SMEM_BYTES, stream>>>(
      atoms, Dn, Eidx, resi, chain, w_pos, b_pos, wperm, gamma, beta, Eout);
}
